// DrugInteractionGCN_52458730553561
// MI455X (gfx1250) — matrix operand order repaired and checked by reading, not compiled, not run on hardware
//
#include <hip/hip_runtime.h>

typedef __attribute__((ext_vector_type(16))) _Float16 v16h;
typedef __attribute__((ext_vector_type(8)))  float    v8f;

#define BN_EPS 1e-5f

// ------------------------------------------------------------------ degree
__global__ void k_fill1(float* p, int n) {
    int i = blockIdx.x * blockDim.x + threadIdx.x;
    if (i < n) p[i] = 1.0f;
}
__global__ void k_deg_count(const int* __restrict__ dst, float* __restrict__ deg, int E) {
    int i = blockIdx.x * blockDim.x + threadIdx.x;
    if (i < E) atomicAdd(&deg[dst[i]], 1.0f);
}
__global__ void k_rsqrt(float* p, int n) {
    int i = blockIdx.x * blockDim.x + threadIdx.x;
    if (i < n) p[i] = rsqrtf(p[i]);
}

// ------------------------------------------------------------------ packing
// Packed-A layout (K mult of 32, R mult of 16):
//   idx = ((mt*KS + ks)*32 + lane)*16 + e
//   row = mt*16 + (lane&15);  k = ks*32 + (lane>>4)*8 + (e<8 ? e : e+8)
__global__ void k_pack_a(const float* __restrict__ X, _Float16* __restrict__ PA,
                         int R, int K) {
    long long i = (long long)blockIdx.x * blockDim.x + threadIdx.x;
    long long total = (long long)R * K;
    if (i >= total) return;
    int KS = K >> 5;
    int e = (int)(i & 15);
    int l = (int)((i >> 4) & 31);
    long long rest = i >> 9;
    int ks = (int)(rest % KS);
    int mt = (int)(rest / KS);
    int row = mt * 16 + (l & 15);
    int k   = ks * 32 + ((l >> 4) * 8) + (e < 8 ? e : e + 8);
    PA[i] = (_Float16)X[(long long)row * K + k];
}

// Packed-B layout: idx = ((nt*KS + ks)*32 + lane)*16 + e
//   col = nt*16 + (lane&15);  k = ks*32 + (lane>>4)*8 + (e<8 ? e : e+8)   (W is K x C row-major)
__global__ void k_pack_b(const float* __restrict__ W, _Float16* __restrict__ PB,
                         int K, int C) {
    long long i = (long long)blockIdx.x * blockDim.x + threadIdx.x;
    long long total = (long long)K * C;
    if (i >= total) return;
    int KS = K >> 5;
    int e = (int)(i & 15);
    int l = (int)((i >> 4) & 31);
    long long rest = i >> 9;
    int ks = (int)(rest % KS);
    int nt = (int)(rest / KS);
    int col = nt * 16 + (l & 15);
    int k   = ks * 32 + ((l >> 4) * 8) + (e < 8 ? e : e + 8);
    PB[i] = (_Float16)W[(long long)k * C + col];
}

// pair = concat(z[s], z[t]) packed as A (K = 256, z has 128 features)
__global__ void k_pack_pair(const float* __restrict__ z,
                            const int* __restrict__ sidx, const int* __restrict__ tidx,
                            _Float16* __restrict__ PA, int P) {
    long long i = (long long)blockIdx.x * blockDim.x + threadIdx.x;
    long long total = (long long)P * 256;
    if (i >= total) return;
    const int KS = 8;                    // 256/32
    int e = (int)(i & 15);
    int l = (int)((i >> 4) & 31);
    long long rest = i >> 9;
    int ks = (int)(rest % KS);
    int mt = (int)(rest / KS);
    int row = mt * 16 + (l & 15);
    int k   = ks * 32 + ((l >> 4) * 8) + (e < 8 ? e : e + 8);
    int node, f;
    if (k < 128) { node = sidx[row]; f = k; }
    else         { node = tidx[row]; f = k - 128; }
    PA[i] = (_Float16)z[(long long)node * 128 + f];
}

// BN(eval) + ReLU, writing straight into packed-A layout for next GEMM
__global__ void k_bn_relu_pack(const float* __restrict__ agg,
                               const float* __restrict__ g, const float* __restrict__ be,
                               const float* __restrict__ m, const float* __restrict__ v,
                               _Float16* __restrict__ PA, int R, int K) {
    long long i = (long long)blockIdx.x * blockDim.x + threadIdx.x;
    long long total = (long long)R * K;
    if (i >= total) return;
    int KS = K >> 5;
    int e = (int)(i & 15);
    int l = (int)((i >> 4) & 31);
    long long rest = i >> 9;
    int ks = (int)(rest % KS);
    int mt = (int)(rest / KS);
    int row = mt * 16 + (l & 15);
    int k   = ks * 32 + ((l >> 4) * 8) + (e < 8 ? e : e + 8);
    float x = agg[(long long)row * K + k];
    float val = (x - m[k]) * rsqrtf(v[k] + BN_EPS) * g[k] + be[k];
    val = fmaxf(val, 0.0f);
    PA[i] = (_Float16)val;
}

// ------------------------------------------------------------------ aggregation
// agg = h * dis^2 + b   (self-loop + bias), fully overwrites agg
__global__ void k_init_agg(const float* __restrict__ h, const float* __restrict__ bias,
                           const float* __restrict__ dis, float* __restrict__ agg,
                           int N, int M) {
    long long i = (long long)blockIdx.x * blockDim.x + threadIdx.x;
    long long total = (long long)N * M;
    if (i >= total) return;
    int node = (int)(i / M);
    int f    = (int)(i % M);
    float d = dis[node];
    agg[i] = h[i] * d * d + bias[f];
}

// agg[dst] += h[src] * dis[src]*dis[dst]   (float4 per thread, atomics land in L2)
__global__ void k_scatter(const int* __restrict__ src, const int* __restrict__ dst,
                          const float* __restrict__ dis, const float* __restrict__ h,
                          float* __restrict__ agg, int E, int M) {
    long long i = (long long)blockIdx.x * blockDim.x + threadIdx.x;
    int chunks = M >> 2;
    long long total = (long long)E * chunks;
    if (i >= total) return;
    int e  = (int)(i / chunks);
    int c4 = (int)(i % chunks);
    int s = src[e], d = dst[e];
    float cf = dis[s] * dis[d];
    const float4 hv = *(const float4*)(h + (long long)s * M + c4 * 4);
    float* a = agg + (long long)d * M + c4 * 4;
    atomicAdd(a + 0, hv.x * cf);
    atomicAdd(a + 1, hv.y * cf);
    atomicAdd(a + 2, hv.z * cf);
    atomicAdd(a + 3, hv.w * cf);
}

// ------------------------------------------------------------------ WMMA GEMM
// MODE 0: plain fp32 store.  MODE 1: bias+ReLU, store into packed-A half layout.
// MODE 2: bias+ReLU, fp32 store.
// KS = K/32 (compile-time). Block = 256 threads = 8 waves; wave -> 16 rows x 64 cols.
// B slice (4 n-tiles x full K) staged once in LDS and shared by all 8 waves.
// B tiles are register double-buffered: iteration ks+1's ds_loads issue before
// iteration ks's WMMAs, so the ds wait tolerates the in-flight loads instead of
// serializing LDS latency in front of every matrix op.
template <int MODE, int KS>
__global__ void k_gemm_wmma(const _Float16* __restrict__ PA, const _Float16* __restrict__ PB,
                            const float* __restrict__ bias, float* __restrict__ Cout,
                            _Float16* __restrict__ Pout, int Rtiles, int Ctot) {
    __shared__ _Float16 sB[4 * KS * 512];

    int tid  = threadIdx.x;
    int wave = tid >> 5;
    int lane = tid & 31;
    int nt0  = blockIdx.y * 4;

    // cooperative, coalesced copy of this block's B slice into LDS
    {
        const float4* srcB = (const float4*)(PB + (size_t)nt0 * KS * 512);
        float4*       dstB = (float4*)sB;
        const int chunks = 4 * KS * 64;           // float4 chunks (8 halves each)
#pragma unroll
        for (int c = tid; c < chunks; c += 256) dstB[c] = srcB[c];
    }
    __syncthreads();

    int mt = blockIdx.x * 8 + wave;
    if (mt >= Rtiles) return;                     // wave-uniform, after the barrier

    // preload the wave's full A strip (16 rows x K) into registers
    const v16h* pa = (const v16h*)PA + (size_t)mt * KS * 32 + lane;
    v16h a[KS];
#pragma unroll
    for (int ks = 0; ks < KS; ++ks) a[ks] = pa[ks * 32];

    const v16h* sb = (const v16h*)sB + lane;
    v8f acc[4] = {};

    v16h Bb[2][4];
#pragma unroll
    for (int t = 0; t < 4; ++t) Bb[0][t] = sb[(t * KS) * 32];

#pragma unroll
    for (int ks = 0; ks < KS; ++ks) {
        const int cur = ks & 1, nxt = cur ^ 1;
        if (ks + 1 < KS) {
#pragma unroll
            for (int t = 0; t < 4; ++t) Bb[nxt][t] = sb[(t * KS + ks + 1) * 32];
        }
#pragma unroll
        for (int t = 0; t < 4; ++t)
            acc[t] = __builtin_amdgcn_wmma_f32_16x16x32_f16(
                false, a[ks], false, Bb[cur][t], (short)0, acc[t], false, false);
    }

    int colL  = lane & 15;
    int rbase = (lane >> 4) * 8;
#pragma unroll
    for (int t = 0; t < 4; ++t) {
        int col = (nt0 + t) * 16 + colL;
        float bv = (MODE != 0) ? bias[col] : 0.0f;
#pragma unroll
        for (int r = 0; r < 8; ++r) {
            int row = mt * 16 + rbase + r;
            float vv = acc[t][r] + bv;
            if (MODE != 0) vv = fmaxf(vv, 0.0f);
            if (MODE == 1) {
                // store into packed-A layout of the NEXT gemm (Kn = Ctot)
                int mt2 = row >> 4, rl = row & 15;
                int ks2 = col >> 5, krem = col & 31;
                int hi = (krem >> 3) & 1;
                int e  = (krem & 7) + ((krem >= 16) ? 8 : 0);
                int l2 = rl + hi * 16;
                long long idx = (((long long)mt2 * (Ctot >> 5) + ks2) * 32 + l2) * 16 + e;
                Pout[idx] = (_Float16)vv;
            } else {
                Cout[(long long)row * Ctot + col] = vv;
            }
        }
    }
}

// ------------------------------------------------------------------ final 64 -> 1
__global__ void k_dot64(const float* __restrict__ U, const float* __restrict__ w,
                        const float* __restrict__ b, float* __restrict__ out, int P) {
    int p = blockIdx.x * blockDim.x + threadIdx.x;
    if (p >= P) return;
    const float4* u  = (const float4*)(U + (long long)p * 64);
    const float4* w4 = (const float4*)w;
    float s = 0.0f;
#pragma unroll
    for (int j = 0; j < 16; ++j) {
        float4 a = u[j], c = w4[j];
        s += a.x * c.x + a.y * c.y + a.z * c.z + a.w * c.w;
    }
    out[p] = s + b[0];
}

// ------------------------------------------------------------------ launch
static inline int blocksFor(long long total, int bs) { return (int)((total + bs - 1) / bs); }

extern "C" void kernel_launch(void* const* d_in, const int* in_sizes, int n_in,
                              void* d_out, int out_size, void* d_ws, size_t ws_size,
                              hipStream_t stream) {
    const float* x    = (const float*)d_in[0];
    const int*   ei   = (const int*)d_in[1];
    const int*   eli  = (const int*)d_in[2];
    const float* W1 = (const float*)d_in[3];  const float* b1 = (const float*)d_in[4];
    const float* W2 = (const float*)d_in[5];  const float* b2 = (const float*)d_in[6];
    const float* W3 = (const float*)d_in[7];  const float* b3 = (const float*)d_in[8];
    const float* g1 = (const float*)d_in[9];  const float* be1 = (const float*)d_in[10];
    const float* m1 = (const float*)d_in[11]; const float* v1  = (const float*)d_in[12];
    const float* g2 = (const float*)d_in[13]; const float* be2 = (const float*)d_in[14];
    const float* m2 = (const float*)d_in[15]; const float* v2  = (const float*)d_in[16];
    const float* D1w = (const float*)d_in[17]; const float* D1b = (const float*)d_in[18];
    const float* D2w = (const float*)d_in[19]; const float* D2b = (const float*)d_in[20];
    const float* D3w = (const float*)d_in[21]; const float* D3b = (const float*)d_in[22];
    const float* D4w = (const float*)d_in[23]; const float* D4b = (const float*)d_in[24];

    const int N = in_sizes[0] / 128;
    const int E = in_sizes[1] / 2;
    const int P = in_sizes[2] / 2;
    const int DIN = 128, H = 256, EMB = 128;

    // ---- workspace carving ----
    char* w = (char*)d_ws;
    auto alloc = [&](size_t bytes) -> void* {
        void* p = (void*)w;
        w += (bytes + 255) & ~(size_t)255;
        return p;
    };
    float*     dis  = (float*)alloc((size_t)N * 4);
    _Float16*  PA   = (_Float16*)alloc((size_t)P * 256 * 2);  // node packs + pair pack
    _Float16*  PU1  = (_Float16*)alloc((size_t)P * 256 * 2);  // decoder u1 packed
    float*     hbuf = (float*)alloc((size_t)N * 256 * 4);
    float*     agg  = (float*)alloc((size_t)N * 256 * 4);
    _Float16*  PB1  = (_Float16*)alloc((size_t)128 * 256 * 2);
    _Float16*  PB2  = (_Float16*)alloc((size_t)256 * 256 * 2);
    _Float16*  PB3  = (_Float16*)alloc((size_t)256 * 128 * 2);
    _Float16*  PD1  = (_Float16*)alloc((size_t)256 * 256 * 2);
    _Float16*  PD2  = (_Float16*)alloc((size_t)256 * 128 * 2);
    _Float16*  PD3  = (_Float16*)alloc((size_t)128 * 64 * 2);
    _Float16*  PU2  = PA;            // D2 output (PA dead after D1 reads it)
    float*     u3   = (float*)PU1;   // D3 fp32 output (PU1 dead after D2 reads it)

    const int BS = 256;
    const int*  src = ei;        const int*  dst = ei + E;
    const int*  ps  = eli;       const int*  pt  = eli + P;

    // ---- degree / dis ----
    k_fill1<<<blocksFor(N, BS), BS, 0, stream>>>(dis, N);
    k_deg_count<<<blocksFor(E, BS), BS, 0, stream>>>(dst, dis, E);
    k_rsqrt<<<blocksFor(N, BS), BS, 0, stream>>>(dis, N);

    // ---- weight packing (tiny) ----
    k_pack_b<<<blocksFor((long long)DIN * H, BS), BS, 0, stream>>>(W1,  PB1, DIN, H);
    k_pack_b<<<blocksFor((long long)H * H,   BS), BS, 0, stream>>>(W2,  PB2, H, H);
    k_pack_b<<<blocksFor((long long)H * EMB, BS), BS, 0, stream>>>(W3,  PB3, H, EMB);
    k_pack_b<<<blocksFor((long long)256*256, BS), BS, 0, stream>>>(D1w, PD1, 256, 256);
    k_pack_b<<<blocksFor((long long)256*128, BS), BS, 0, stream>>>(D2w, PD2, 256, 128);
    k_pack_b<<<blocksFor((long long)128*64,  BS), BS, 0, stream>>>(D3w, PD3, 128, 64);

    const int RtN = N / 16, RtP = P / 16;
    dim3 blk(BS);

    // ---- GCN layer 1 ----
    k_pack_a<<<blocksFor((long long)N * DIN, BS), BS, 0, stream>>>(x, PA, N, DIN);
    k_gemm_wmma<0, 4><<<dim3((RtN + 7) / 8, H / 64), blk, 0, stream>>>(PA, PB1, nullptr, hbuf, nullptr, RtN, H);
    k_init_agg<<<blocksFor((long long)N * H, BS), BS, 0, stream>>>(hbuf, b1, dis, agg, N, H);
    k_scatter<<<blocksFor((long long)E * (H / 4), BS), BS, 0, stream>>>(src, dst, dis, hbuf, agg, E, H);
    k_bn_relu_pack<<<blocksFor((long long)N * H, BS), BS, 0, stream>>>(agg, g1, be1, m1, v1, PA, N, H);

    // ---- GCN layer 2 ----
    k_gemm_wmma<0, 8><<<dim3((RtN + 7) / 8, H / 64), blk, 0, stream>>>(PA, PB2, nullptr, hbuf, nullptr, RtN, H);
    k_init_agg<<<blocksFor((long long)N * H, BS), BS, 0, stream>>>(hbuf, b2, dis, agg, N, H);
    k_scatter<<<blocksFor((long long)E * (H / 4), BS), BS, 0, stream>>>(src, dst, dis, hbuf, agg, E, H);
    k_bn_relu_pack<<<blocksFor((long long)N * H, BS), BS, 0, stream>>>(agg, g2, be2, m2, v2, PA, N, H);

    // ---- GCN layer 3 (no BN/ReLU) -> z = agg (N x 128) ----
    k_gemm_wmma<0, 8><<<dim3((RtN + 7) / 8, EMB / 64), blk, 0, stream>>>(PA, PB3, nullptr, hbuf, nullptr, RtN, EMB);
    k_init_agg<<<blocksFor((long long)N * EMB, BS), BS, 0, stream>>>(hbuf, b3, dis, agg, N, EMB);
    k_scatter<<<blocksFor((long long)E * (EMB / 4), BS), BS, 0, stream>>>(src, dst, dis, hbuf, agg, E, EMB);

    // ---- decoder ----
    k_pack_pair<<<blocksFor((long long)P * 256, BS), BS, 0, stream>>>(agg, ps, pt, PA, P);
    k_gemm_wmma<1, 8><<<dim3((RtP + 7) / 8, 256 / 64), blk, 0, stream>>>(PA,  PD1, D1b, nullptr, PU1, RtP, 256);
    k_gemm_wmma<1, 8><<<dim3((RtP + 7) / 8, 128 / 64), blk, 0, stream>>>(PU1, PD2, D2b, nullptr, PU2, RtP, 128);
    k_gemm_wmma<2, 4><<<dim3((RtP + 7) / 8, 64 / 64),  blk, 0, stream>>>(PU2, PD3, D3b, u3, nullptr, RtP, 64);
    k_dot64<<<blocksFor(P, BS), BS, 0, stream>>>(u3, D4w, D4b, (float*)d_out, P);
}
